// PottsTransferModel_87514253623854
// MI455X (gfx1250) — matrix operand order repaired and checked by reading, not compiled, not run on hardware
//
#include <hip/hip_runtime.h>
#include <hip/hip_bf16.h>

// ---------------------------------------------------------------------------
// PottsTransferModel forward, optimized for MI455X (gfx1250, wave32, WMMA).
// lin has <=30 nonzero rows/batch; masked softmax weights underflow to exact
// zero elsewhere, so conv + MLP run only on neighbor columns (+1 zero row
// that yields the constant output row). GEMMs: v_wmma_f32_16x16x32_f16.
// All fragments are loaded as contiguous 16B vectors straight from
// global memory (weights pre-packed f16, lin stored f16) -- no LDS staging,
// no per-element guards.
// ---------------------------------------------------------------------------

typedef __attribute__((ext_vector_type(16))) _Float16 v16h;
typedef __attribute__((ext_vector_type(8)))  _Float16 hv8;
typedef __attribute__((ext_vector_type(8)))  float    v8f;
typedef __attribute__((ext_vector_type(4)))  float    f4;

#define BB    8
#define NN    512
#define KNB   30
#define HAA   20
#define HPAD  22
#define EMB   484
#define CPAD  512     // padded channel dim (32-wide K chunks)
#define OCP   496     // padded out-channel dim (31 tiles of 16)
#define KSZ   9
#define SMUT  99
#define ROWS  32      // compact-position slots per batch

static __device__ __forceinline__ v16h cat8(hv8 lo, hv8 hi) {
  return __builtin_shufflevector(lo, hi, 0, 1, 2, 3, 4, 5, 6, 7,
                                         8, 9, 10, 11, 12, 13, 14, 15);
}

// ---------------------------------------------------------------------------
// 0a) Pack conv weights to f16: Wc[cv][oc 496][t 9][ic 512]
// ---------------------------------------------------------------------------
__global__ void pack_conv_w(const float* __restrict__ Wf,
                            const float* __restrict__ Wa,
                            _Float16* __restrict__ Wc) {
  size_t idx = (size_t)blockIdx.x * blockDim.x + threadIdx.x; // 2*496*9*512
  int ic = idx & (CPAD - 1);
  size_t rest = idx >> 9;
  int t = rest % KSZ; rest /= KSZ;
  int oc = rest % OCP;
  int cv = rest / OCP;
  const float* W = cv ? Wa : Wf;
  float v = (oc < EMB && ic < EMB) ? W[((size_t)oc * EMB + ic) * KSZ + t] : 0.0f;
  Wc[idx] = (_Float16)v;
}

// ---------------------------------------------------------------------------
// 0b) Pack dense weights to f16: Wd[layer][oc 512][ic 512]
// ---------------------------------------------------------------------------
__global__ void pack_dense_w(const float* __restrict__ W1,
                             const float* __restrict__ W2,
                             _Float16* __restrict__ Wd) {
  size_t idx = (size_t)blockIdx.x * blockDim.x + threadIdx.x; // 2*512*512
  int ic = idx & (CPAD - 1);
  int oc = (idx >> 9) & (CPAD - 1);
  int layer = idx >> 18;
  const float* W = layer ? W2 : W1;
  float v = (oc < EMB && ic < EMB) ? W[(size_t)oc * EMB + ic] : 0.0f;
  Wd[idx] = (_Float16)v;
}

// ---------------------------------------------------------------------------
// 1) Scatter: build lin_h[b][n][512] (f16, padded 20x20 -> 22x22 rows at
//    neighbor positions), dedup neighbor list, build pos2idx map.
// ---------------------------------------------------------------------------
__global__ void scatter_kernel(const float* __restrict__ etab,
                               const int* __restrict__ E_idx,
                               const int* __restrict__ mut_pos,
                               _Float16* __restrict__ lin_h,
                               int* __restrict__ poslist,
                               int* __restrict__ m_count,
                               int* __restrict__ pos2idx) {
  int b = blockIdx.x;
  int tid = threadIdx.x;
  __shared__ int nbrs_s[KNB];
  int mp = mut_pos[b];
  if (tid == 0) {
    int m = 0;
    for (int kk = 0; kk < KNB; ++kk) {
      int nbr = E_idx[((b * NN) + mp) * KNB + kk];
      nbrs_s[kk] = nbr;
      if (pos2idx[b * NN + nbr] < 0) {
        pos2idx[b * NN + nbr] = m;
        poslist[b * ROWS + m] = nbr;
        m++;
      }
    }
    m_count[b] = m;
    for (int r = m; r < ROWS; ++r) poslist[b * ROWS + r] = -1;
  }
  __syncthreads();
  // serial over kk with barriers => last-write-wins for duplicate neighbors
  for (int kk = 0; kk < KNB; ++kk) {
    int nbr = nbrs_s[kk];
    const float* src = etab + (size_t)(((b * NN) + mp) * KNB + kk) * (HAA * HAA);
    _Float16* dst = lin_h + (size_t)(b * NN + nbr) * CPAD;
    if (tid < EMB) {
      int a = tid / HPAD, c = tid % HPAD;
      dst[tid] = (_Float16)((a < HAA && c < HAA) ? src[a * HAA + c] : 0.0f);
    }
    __syncthreads();
  }
}

// ---------------------------------------------------------------------------
// 2) Conv-as-GEMM at neighbor columns. Block = 16 out-chans x 32 positions
//    (2 waves x 16). All fragment data loaded directly as 16B vectors.
// ---------------------------------------------------------------------------
__global__ void conv_kernel(const _Float16* __restrict__ lin_h,
                            const _Float16* __restrict__ Wc,
                            const float* __restrict__ bf,
                            const float* __restrict__ ba,
                            const int* __restrict__ poslist,
                            float* __restrict__ oatt) {
  int mt = blockIdx.x;            // out-channel tile 0..30
  int b  = blockIdx.y;            // batch
  int cv = blockIdx.z;            // 0 = o (W_f), 1 = att (W_a)
  int tid  = threadIdx.x;         // 0..63
  int wv   = tid >> 5;            // wave id = position tile
  int lane = tid & 31;
  int rowL = lane & 15;
  int hiK  = lane >> 4;
  int m0 = mt * 16;

  // per-lane A row: Wc[cv][m0+rowL][t][*]
  const _Float16* arow = Wc + ((size_t)(cv * OCP + m0 + rowL)) * KSZ * CPAD;
  // per-lane B column position
  int l = poslist[b * ROWS + wv * 16 + rowL];
  const _Float16* lbase = lin_h + (size_t)b * NN * CPAD;
  const _Float16* zrow  = lin_h + (size_t)BB * NN * CPAD;   // dedicated zero row

  v8f acc = {};
  for (int t = 0; t < KSZ; ++t) {
    int r = l + t - 4;
    bool bvalid = (l >= 0) & (r >= 0) & (r < NN);
    const _Float16* brow = bvalid ? (lbase + (size_t)r * CPAD) : zrow;
    const _Float16* at = arow + t * CPAD;
#pragma unroll 4
    for (int ci = 0; ci < 16; ++ci) {
      int c0 = ci * 32;
      hv8 a0 = *(const hv8*)(at + c0 + hiK * 8);
      hv8 a1 = *(const hv8*)(at + c0 + hiK * 8 + 16);
      hv8 b0 = *(const hv8*)(brow + c0 + hiK * 8);
      hv8 b1 = *(const hv8*)(brow + c0 + hiK * 8 + 16);
      acc = __builtin_amdgcn_wmma_f32_16x16x32_f16(false, cat8(a0, a1),
                                                   false, cat8(b0, b1),
                                                   (short)0, acc, false, false);
    }
  }
  // store transposed: oatt[b][cv][pos][chan]
  const float* bias = cv ? ba : bf;
  int pos = wv * 16 + rowL;
#pragma unroll
  for (int v = 0; v < 8; ++v) {
    int oc = m0 + v + 8 * hiK;
    float bv = (oc < EMB) ? bias[oc] : 0.0f;
    oatt[(((size_t)(b * 2 + cv)) * ROWS + pos) * CPAD + oc] = acc[v] + bv;
  }
}

// ---------------------------------------------------------------------------
// 3) Per-channel softmax over the m valid positions; la = o * weight.
//    Rows [m,32) / chans [484,512) written as zeros (zero-input rows give
//    the constant MLP row for free).
// ---------------------------------------------------------------------------
__global__ void softmax_kernel(const float* __restrict__ oatt,
                               const int* __restrict__ m_count,
                               float* __restrict__ la) {
  int b = blockIdx.x;
  int c = threadIdx.x;  // 0..511
  int m = m_count[b];
  const float* o  = oatt + ((size_t)(b * 2 + 0)) * ROWS * CPAD;
  const float* at = oatt + ((size_t)(b * 2 + 1)) * ROWS * CPAD;
  if (c < EMB) {
    float mx = -1e30f;
    for (int i = 0; i < m; ++i) mx = fmaxf(mx, at[i * CPAD + c]);
    float Z = 0.0f;
    for (int i = 0; i < m; ++i) Z += expf(at[i * CPAD + c] - mx);
    float inv = 1.0f / Z;
    for (int i = 0; i < ROWS; ++i) {
      float v = (i < m) ? o[i * CPAD + c] * expf(at[i * CPAD + c] - mx) * inv : 0.0f;
      la[((size_t)(b * ROWS + i)) * CPAD + c] = v;
    }
  } else {
    for (int i = 0; i < ROWS; ++i)
      la[((size_t)(b * ROWS + i)) * CPAD + c] = 0.0f;
  }
}

// ---------------------------------------------------------------------------
// 4) Dense layer: Y = relu(X) @ W^T + bias. 256 rows x 512 out, K=512.
//    No LDS; A = four float4 loads + fused ReLU/cvt, B = two b128 f16 loads.
// ---------------------------------------------------------------------------
__global__ void dense_kernel(const float* __restrict__ X,
                             const _Float16* __restrict__ Wd,
                             const float* __restrict__ bias,
                             float* __restrict__ Y) {
  int mt = blockIdx.x;   // row tile 0..15
  int np = blockIdx.y;   // N pair 0..15
  int tid = threadIdx.x;
  int wv = tid >> 5, lane = tid & 31, rowL = lane & 15, hiK = lane >> 4;
  int r0 = mt * 16, n0 = np * 32;
  int nc = n0 + wv * 16 + rowL;
  const float*    xrow = X  + (size_t)(r0 + rowL) * CPAD;
  const _Float16* wrow = Wd + (size_t)nc * CPAD;

  v8f acc = {};
#pragma unroll 4
  for (int ci = 0; ci < 16; ++ci) {
    int k0 = ci * 32;
    int ka = k0 + hiK * 8;
    f4 a0 = *(const f4*)(xrow + ka);
    f4 a1 = *(const f4*)(xrow + ka + 4);
    f4 a2 = *(const f4*)(xrow + ka + 16);
    f4 a3 = *(const f4*)(xrow + ka + 20);
    v16h af;
#pragma unroll
    for (int i = 0; i < 4; ++i) {
      af[i]      = (_Float16)fmaxf(a0[i], 0.0f);
      af[4 + i]  = (_Float16)fmaxf(a1[i], 0.0f);
      af[8 + i]  = (_Float16)fmaxf(a2[i], 0.0f);
      af[12 + i] = (_Float16)fmaxf(a3[i], 0.0f);
    }
    hv8 b0 = *(const hv8*)(wrow + k0 + hiK * 8);
    hv8 b1 = *(const hv8*)(wrow + k0 + hiK * 8 + 16);
    acc = __builtin_amdgcn_wmma_f32_16x16x32_f16(false, af,
                                                 false, cat8(b0, b1),
                                                 (short)0, acc, false, false);
  }
#pragma unroll
  for (int v = 0; v < 8; ++v) {
    int r = r0 + v + 8 * hiK;
    float bv = (nc < EMB) ? bias[nc] : 0.0f;
    Y[((size_t)r) * CPAD + nc] = acc[v] + bv;
  }
}

// ---------------------------------------------------------------------------
// 5) Scoring. score[b,s] = sum_j y2row(b,j)[spos*22 + seq_j]; non-neighbor j
//    uses the constant row (slot 31, guaranteed zero-input). b_ddg cancels in
//    the WT difference; w_ddg factors out.
// ---------------------------------------------------------------------------
__global__ void score_kernel(const float* __restrict__ y2,
                             const int* __restrict__ pos2idx,
                             const int* __restrict__ sortcery,
                             const int* __restrict__ seqs,
                             const int* __restrict__ mut_pos,
                             const float* __restrict__ w_ddg,
                             float* __restrict__ out) {
  int b = blockIdx.x;
  int s = blockIdx.y;      // 0..98
  int tid = threadIdx.x;   // 0..255
  int mp = mut_pos[b];
  const int* sq = sortcery + ((size_t)b * SMUT + s) * NN;
  const int* wq = seqs + (size_t)b * NN;
  int spos = sq[mp];
  int wpos = wq[mp];
  float accS = 0.0f, accW = 0.0f;
  for (int j = tid; j < NN; j += 256) {
    int idx = pos2idx[b * NN + j];
    int r = (idx >= 0) ? idx : (ROWS - 1);
    const float* row = y2 + ((size_t)(b * ROWS + r)) * CPAD;
    accS += row[spos * HPAD + sq[j]];
    accW += row[wpos * HPAD + wq[j]];
  }
  __shared__ float sS[256], sW[256];
  sS[tid] = accS; sW[tid] = accW;
  __syncthreads();
  for (int st = 128; st > 0; st >>= 1) {
    if (tid < st) { sS[tid] += sS[tid + st]; sW[tid] += sW[tid + st]; }
    __syncthreads();
  }
  if (tid == 0) out[b * SMUT + s] = w_ddg[0] * (sS[0] - sW[0]);
}

// ---------------------------------------------------------------------------
extern "C" void kernel_launch(void* const* d_in, const int* in_sizes, int n_in,
                              void* d_out, int out_size, void* d_ws, size_t ws_size,
                              hipStream_t stream) {
  const float* etab     = (const float*)d_in[0];
  const int*   E_idx    = (const int*)  d_in[1];
  const int*   sortcery = (const int*)  d_in[2];
  const int*   seqs     = (const int*)  d_in[3];
  const int*   mut_pos  = (const int*)  d_in[4];
  // d_in[5] = nodes (unused by the reference math)
  const float* Wf = (const float*)d_in[6];
  const float* bf = (const float*)d_in[7];
  const float* Wa = (const float*)d_in[8];
  const float* ba = (const float*)d_in[9];
  const float* W1 = (const float*)d_in[10];
  const float* b1 = (const float*)d_in[11];
  const float* W2 = (const float*)d_in[12];
  const float* b2 = (const float*)d_in[13];
  const float* w_ddg = (const float*)d_in[14];
  // d_in[15] = b_ddg: cancels in WT-subtracted scores

  char* ws = (char*)d_ws;
  _Float16* lin_h = (_Float16*)(ws + 0);         // (8*512+1)*512*2 = 4,195,328
  _Float16* Wc    = (_Float16*)(ws + 4195328);   // 2*496*9*512*2   = 9,142,272
  _Float16* Wd    = (_Float16*)(ws + 13337600);  // 2*512*512*2     = 1,048,576
  float* oatt     = (float*)   (ws + 14386176);  // 8*2*32*512*4    = 1,048,576
  float* la       = (float*)   (ws + 15434752);  // 256*512*4       =   524,288
  float* y1       = (float*)   (ws + 15959040);  //                     524,288
  float* y2       = (float*)   (ws + 16483328);  //                     524,288
  int* poslist    = (int*)     (ws + 17007616);  // 8*32*4
  int* m_count    = (int*)     (ws + 17008640);  // 8*4 (padded)
  int* pos2idx    = (int*)     (ws + 17008896);  // 8*512*4

  // lin_h (incl. trailing zero row) = 0; pos2idx = -1
  hipMemsetAsync(lin_h, 0, ((size_t)BB * NN + 1) * CPAD * sizeof(_Float16), stream);
  hipMemsetAsync(pos2idx, 0xFF, (size_t)BB * NN * sizeof(int), stream);

  pack_conv_w<<<(2 * OCP * KSZ * CPAD) / 256, 256, 0, stream>>>(Wf, Wa, Wc);
  pack_dense_w<<<(2 * CPAD * CPAD) / 256, 256, 0, stream>>>(W1, W2, Wd);

  scatter_kernel<<<BB, 512, 0, stream>>>(etab, E_idx, mut_pos, lin_h,
                                         poslist, m_count, pos2idx);
  conv_kernel<<<dim3(31, BB, 2), 64, 0, stream>>>(lin_h, Wc, bf, ba,
                                                  poslist, oatt);
  softmax_kernel<<<BB, 512, 0, stream>>>(oatt, m_count, la);
  dense_kernel<<<dim3(16, 16), 64, 0, stream>>>(la, Wd, b1, y1);
  dense_kernel<<<dim3(16, 16), 64, 0, stream>>>(y1, Wd + (size_t)CPAD * CPAD, b2, y2);
  score_kernel<<<dim3(BB, SMUT), 256, 0, stream>>>(y2, pos2idx, sortcery, seqs,
                                                   mut_pos, w_ddg, (float*)d_out);
}
